// ContextSNN_42004780154961
// MI455X (gfx1250) — matrix operand order repaired and checked by reading, not compiled, not run on hardware
//
#include <hip/hip_runtime.h>

typedef __attribute__((ext_vector_type(16))) _Float16 v16h;
typedef __attribute__((ext_vector_type(4)))  _Float16 v4h;
typedef __attribute__((ext_vector_type(8)))  float    v8f;

#define T_STEPS 50
#define BATCH   4096
#define IN_F    1500
#define HID     512
#define OUT_F   45
#define BETA    0.9f
#define THR     1.0f

#define IN_PAD  1536
#define KT1     48          // IN_PAD / 32 k-tiles for GEMM1
#define NCHUNK  12          // K chunks of 128 (4 k-tiles) per timestep
#define H_KT    16          // HID / 32 k-tiles for GEMM2
#define NT1     32          // HID / 16 n-tiles
#define NT2     3           // ceil(OUT_F / 16) n-tiles (padded to 48)

#define BM      64          // batch rows per block (4 M-tiles)
#define NBLK    (BATCH / BM)

#define W1P_HALVES (NT1 * KT1 * 512)   // 786432 halves = 1.5 MB
#define W2P_HALVES (NT2 * H_KT * 512)  // 24576 halves  = 48 KB

// ---- WMMA fragment layout helpers (wave32, 16x16x32 f16) ----
// A (16x32, 16 halves/lane): lane L holds row M = L&15;
//   half j -> K = ((j>>3)<<4) + ((L>>4)<<3) + (j&7)
// B (32x16, 16 halves/lane): lane L holds col N = L&15;
//   half j -> K = ((j>>3)<<4) + ((L>>4)<<3) + (j&7)
// C/D (16x16 f32, 8 VGPRs): vgpr r, lane L -> M = r + ((L>>4)<<3), N = L&15

__global__ void pack_w1(const float* __restrict__ W1, _Float16* __restrict__ W1p) {
    int idx = blockIdx.x * 256 + threadIdx.x;          // one thread per half
    if (idx >= W1P_HALVES) return;
    int j    = idx & 15;
    int lane = (idx >> 4) & 31;
    int frag = idx >> 9;
    int kt   = frag % KT1;
    int nt   = frag / KT1;
    int n = nt * 16 + (lane & 15);
    int k = kt * 32 + ((j >> 3) << 4) + ((lane >> 4) << 3) + (j & 7);
    float v = (k < IN_F) ? W1[(size_t)n * IN_F + k] : 0.f;
    W1p[idx] = (_Float16)v;
}

__global__ void pack_w2(const float* __restrict__ W2, _Float16* __restrict__ W2p) {
    int idx = blockIdx.x * 256 + threadIdx.x;
    if (idx >= W2P_HALVES) return;
    int j    = idx & 15;
    int lane = (idx >> 4) & 31;
    int frag = idx >> 9;
    int kt   = frag % H_KT;
    int nt   = frag / H_KT;
    int n = nt * 16 + (lane & 15);
    int k = kt * 32 + ((j >> 3) << 4) + ((lane >> 4) << 3) + (j & 7);
    float v = (n < OUT_F) ? W2[(size_t)n * HID + k] : 0.f;
    W2p[idx] = (_Float16)v;
}

__global__ __launch_bounds__(256)
void snn_kernel(const float* __restrict__ xseq,
                const _Float16* __restrict__ W1p,
                const float* __restrict__ b1,
                const _Float16* __restrict__ W2p,
                const float* __restrict__ b2,
                float* __restrict__ out)
{
    extern __shared__ _Float16 smem[];
    _Float16* xbuf = smem;            // [4 mtiles][4 kt][32 lanes][16] = 8192 halves
    _Float16* sbuf = smem + 8192;     // [4 mtiles][16 kt][32 lanes][16] = 32768 halves

    const int tid  = threadIdx.x;
    const int lane = tid & 31;
    const int wave = tid >> 5;        // 0..7
    const int row0 = blockIdx.x * BM;
    const int n_lo = lane & 15;
    const int l_hi = lane >> 4;

    // Per-lane bias values (C-layout column = lane&15, same for all 8 vgprs)
    float b1v[4];
    #pragma unroll
    for (int ni = 0; ni < 4; ++ni)
        b1v[ni] = b1[(wave * 4 + ni) * 16 + n_lo];
    float b2v[NT2];
    #pragma unroll
    for (int ni = 0; ni < NT2; ++ni) {
        int c = ni * 16 + n_lo;
        b2v[ni] = (c < OUT_F) ? b2[c] : 0.f;
    }

    // Persistent state in WMMA accumulator registers
    v8f mem1[4][4];                   // [mtile][ntile] of hidden membrane
    #pragma unroll
    for (int mi = 0; mi < 4; ++mi)
        #pragma unroll
        for (int ni = 0; ni < 4; ++ni)
            #pragma unroll
            for (int r = 0; r < 8; ++r) mem1[mi][ni][r] = 0.f;

    v8f mem2[NT2], acc2[NT2];         // waves 0..3 only: mtile = wave
    #pragma unroll
    for (int ni = 0; ni < NT2; ++ni)
        #pragma unroll
        for (int r = 0; r < 8; ++r) { mem2[ni][r] = 0.f; acc2[ni][r] = 0.f; }

    // Staging geometry (constant per thread)
    const int st_rl    = tid >> 2;             // local row 0..63
    const int st_ks    = (tid & 3) * 32;       // 32-wide K segment within chunk
    const int st_mtile = st_rl >> 4;
    const int st_m     = st_rl & 15;

    for (int t = 0; t < T_STEPS; ++t) {
        // ---- LIF1 pre-update: acc = beta*mem - rst*thr + b1 (WMMA adds x@W1^T) ----
        #pragma unroll
        for (int mi = 0; mi < 4; ++mi)
            #pragma unroll
            for (int ni = 0; ni < 4; ++ni)
                #pragma unroll
                for (int r = 0; r < 8; ++r) {
                    float m = mem1[mi][ni][r];
                    float rst = (m > THR) ? THR : 0.f;
                    mem1[mi][ni][r] = BETA * m - rst + b1v[ni];
                }

        const float* xt   = xseq + ((size_t)t * BATCH + row0) * IN_F;
        const float* xrow = xt + (size_t)st_rl * IN_F;

        // ---- GEMM1: stream K in chunks of 128 through LDS ----
        for (int kc = 0; kc < NCHUNK; ++kc) {
            __syncthreads();   // previous chunk's A-frag reads complete
            {
                // stage + convert + swizzle x chunk into A-fragment layout
                // 4-aligned quad maps to 4 contiguous halves -> one b64 store
                #pragma unroll
                for (int q = 0; q < 8; ++q) {
                    int kl = st_ks + q * 4;        // chunk-local k (mult of 4)
                    int kg = kc * 128 + kl;        // global k
                    v4h pk = (v4h)(_Float16)0;
                    if (kg < IN_F) {               // IN_F % 4 == 0: quad never straddles
                        const float4 f = *(const float4*)(xrow + kg);
                        pk[0] = (_Float16)f.x; pk[1] = (_Float16)f.y;
                        pk[2] = (_Float16)f.z; pk[3] = (_Float16)f.w;
                    }
                    int ktl = kl >> 5;
                    int ko  = kl & 31;
                    int lp  = st_m + (((ko >> 3) & 1) << 4);
                    int jj  = ((ko >> 4) << 3) + (ko & 7);   // jj % 4 == 0
                    *(v4h*)&xbuf[((st_mtile * 4 + ktl) * 32 + lp) * 16 + jj] = pk;
                }
            }
            __syncthreads();   // chunk staged

            #pragma unroll
            for (int kt = 0; kt < 4; ++kt) {
                v16h a[4];
                #pragma unroll
                for (int mi = 0; mi < 4; ++mi)
                    a[mi] = *(const v16h*)&xbuf[((mi * 4 + kt) * 32 + lane) * 16];
                v16h bf[4];
                #pragma unroll
                for (int ni = 0; ni < 4; ++ni) {
                    int nt = wave * 4 + ni;
                    bf[ni] = *(const v16h*)&W1p[(size_t)((nt * KT1 + kc * 4 + kt) * 32 + lane) * 16];
                }
                #pragma unroll
                for (int mi = 0; mi < 4; ++mi)
                    #pragma unroll
                    for (int ni = 0; ni < 4; ++ni)
                        mem1[mi][ni] = __builtin_amdgcn_wmma_f32_16x16x32_f16(
                            false, a[mi], false, bf[ni], (short)0, mem1[mi][ni], false, false);
            }
        }

        // ---- spike1 -> LDS in A-fragment layout (K dim = hidden) ----
        #pragma unroll
        for (int mi = 0; mi < 4; ++mi)
            #pragma unroll
            for (int ni = 0; ni < 4; ++ni) {
                int nt  = wave * 4 + ni;
                int col = nt * 16 + n_lo;          // hidden index
                int kt2 = col >> 5;
                int ko  = col & 31;
                int lpb = ((ko >> 3) & 1) << 4;
                int jj  = ((ko >> 4) << 3) + (ko & 7);
                #pragma unroll
                for (int r = 0; r < 8; ++r) {
                    float s = (mem1[mi][ni][r] - THR > 0.f) ? 1.f : 0.f;
                    int m = r + (l_hi << 3);
                    sbuf[((mi * H_KT + kt2) * 32 + (m + lpb)) * 16 + jj] = (_Float16)s;
                }
            }
        __syncthreads();   // spk1 visible

        // ---- GEMM2 + LIF2 + spike-count accumulate (waves 0..3, mtile = wave) ----
        if (wave < 4) {
            #pragma unroll
            for (int ni = 0; ni < NT2; ++ni)
                #pragma unroll
                for (int r = 0; r < 8; ++r) {
                    float m = mem2[ni][r];
                    float rst = (m > THR) ? THR : 0.f;
                    mem2[ni][r] = BETA * m - rst + b2v[ni];
                }
            #pragma unroll 4
            for (int kt2 = 0; kt2 < H_KT; ++kt2) {
                v16h a = *(const v16h*)&sbuf[((wave * H_KT + kt2) * 32 + lane) * 16];
                #pragma unroll
                for (int ni = 0; ni < NT2; ++ni) {
                    v16h bf = *(const v16h*)&W2p[(size_t)(ni * H_KT + kt2) * 512 + lane * 16];
                    mem2[ni] = __builtin_amdgcn_wmma_f32_16x16x32_f16(
                        false, a, false, bf, (short)0, mem2[ni], false, false);
                }
            }
            #pragma unroll
            for (int ni = 0; ni < NT2; ++ni)
                #pragma unroll
                for (int r = 0; r < 8; ++r)
                    acc2[ni][r] += (mem2[ni][r] - THR > 0.f) ? 1.f : 0.f;
        }
    }

    // ---- write spike counts [B, 45] ----
    if (wave < 4) {
        #pragma unroll
        for (int ni = 0; ni < NT2; ++ni) {
            int col = ni * 16 + n_lo;
            if (col < OUT_F) {
                #pragma unroll
                for (int r = 0; r < 8; ++r) {
                    int row = row0 + wave * 16 + r + (l_hi << 3);
                    out[(size_t)row * OUT_F + col] = acc2[ni][r];
                }
            }
        }
    }
}

extern "C" void kernel_launch(void* const* d_in, const int* in_sizes, int n_in,
                              void* d_out, int out_size, void* d_ws, size_t ws_size,
                              hipStream_t stream) {
    const float* xseq = (const float*)d_in[0];
    const float* W1   = (const float*)d_in[1];
    const float* b1   = (const float*)d_in[2];
    const float* W2   = (const float*)d_in[3];
    const float* b2   = (const float*)d_in[4];
    float* out = (float*)d_out;

    if (ws_size < (size_t)(W1P_HALVES + W2P_HALVES) * sizeof(_Float16)) return;
    _Float16* W1p = (_Float16*)d_ws;
    _Float16* W2p = W1p + W1P_HALVES;

    pack_w1<<<(W1P_HALVES + 255) / 256, 256, 0, stream>>>(W1, W1p);
    pack_w2<<<(W2P_HALVES + 255) / 256, 256, 0, stream>>>(W2, W2p);

    size_t shmem = (size_t)(8192 + 32768) * sizeof(_Float16);  // 80 KB
    snn_kernel<<<NBLK, 256, shmem, stream>>>(xseq, W1p, b1, W2p, b2, out);
}
